// Model2d3d_27109833572949
// MI455X (gfx1250) — matrix operand order repaired and checked by reading, not compiled, not run on hardware
//
#include <hip/hip_runtime.h>
#include <math.h>

typedef __attribute__((ext_vector_type(16))) _Float16 v16h;
typedef __attribute__((ext_vector_type(2)))  _Float16 v2h;
typedef __attribute__((ext_vector_type(8)))  float    v8f;

// ---------------- problem constants (from reference) ----------------
constexpr int   kB    = 2;
constexpr int   kV    = 8;
constexpr int   kBV   = 16;        // B*V
constexpr int   kN    = 8192;      // points
constexpr int   kHW   = 32 * 41;   // 1312 image pixels (multiple of 16)
constexpr int   kCimg = 32;        // conv output channels
constexpr int   kS    = 1024;      // level-0 npoint (only live SA level)
constexpr int   kNCLS = 21;
constexpr float kBNS  = 0.9999950000374997f;  // 1/sqrt(1+1e-5)

static inline int ceilDiv(int a, int b) { return (a + b - 1) / b; }

// =====================================================================
// Weight pre-pack: f32 W[M,K] (BN scale folded) -> f16 WMMA A-fragments,
// layout [mt][kt][lane][16] halves, zero padded to 16x32 tiles.
// =====================================================================
__global__ void pack_w_kernel(const float* __restrict__ W, const float* __restrict__ g,
                              float sm, _Float16* __restrict__ out,
                              int M, int K, int KT, int total)
{
    int t = blockIdx.x * blockDim.x + threadIdx.x;
    if (t >= total) return;
    int j    = t & 15;
    int lane = (t >> 4) & 31;
    int tile = t >> 9;
    int kt = tile % KT, mt = tile / KT;
    int hf = lane >> 4, rc = lane & 15;
    int q = j >> 1, lo = j & 1;
    int kb = hf * 16 + j; (void)q; (void)lo;
    int m = mt * 16 + rc;
    int k = kt * 32 + kb;
    float v = 0.f;
    if (m < M && k < K) {
        v = W[(size_t)m * K + k];
        if (g) v *= g[m] * sm;
    }
    out[t] = (_Float16)v;
}

// =====================================================================
// Branch-free WMMA GEMM: out[M,N] = act( Wpacked @ X + bias ).
// Preconditions: N % 16 == 0; X has KT*32 rows (rows K..KT*32-1 zeroed).
// One wave per block; wave owns MW vertical 16x16 tiles (MW accumulators
// share one staged B tile). B tile is software-pipelined through registers
// and double-buffered in LDS (one barrier per K step); all MW A fragments
// are loaded in one clause before the WMMA burst.
// flags: bit0 ReLU, bit1 transposed store. blockIdx.z batches via strides.
// =====================================================================
template <int MW>
__global__ void __launch_bounds__(32)
gemm_packed(const _Float16* __restrict__ Wp, const float* __restrict__ Xg,
            const float* __restrict__ bias, float* __restrict__ Og,
            int M, int N, int KT, int MT, int flags, long xStride, long oStride)
{
    __shared__ v16h lbF[2][32];              // double-buffered B fragments
    const float* X   = Xg + (long)blockIdx.z * xStride;
    float*       out = Og + (long)blockIdx.z * oStride;

    const int n0   = blockIdx.x * 16;
    const int mt0  = blockIdx.y * MW;
    const int lane = threadIdx.x;
    const int hf   = lane >> 4;
    const int rc   = lane & 15;

    v8f acc[MW] = {};
    const _Float16* wp = Wp + (size_t)mt0 * KT * 512 + (size_t)lane * 16;

    // per-lane staged B elements (8 k-pairs) kept in registers
    float blo[8], bhi[8];
    int colr[8], pr[8];
    #pragma unroll
    for (int i = 0; i < 8; ++i) {
        int pp  = i * 32 + lane;             // k-pair index 0..255
        colr[i] = pp & 15;
        pr[i]   = pp >> 4;                   // 0..15 -> k rows 2p, 2p+1
    }

    // prologue: fetch B tile 0
    #pragma unroll
    for (int i = 0; i < 8; ++i) {
        size_t krow = (size_t)2 * pr[i];
        blo[i] = X[krow * N + n0 + colr[i]];
        bhi[i] = X[(krow + 1) * N + n0 + colr[i]];
    }

    for (int kt = 0; kt < KT; ++kt) {
        // convert + store current B tile into fragment-ordered LDS buffer
        v2h* lb2 = (v2h*)lbF[kt & 1];
        #pragma unroll
        for (int i = 0; i < 8; ++i) {
            v2h h; h[0] = (_Float16)blo[i]; h[1] = (_Float16)bhi[i];
            lb2[((((pr[i] >> 3) << 4) + colr[i]) << 3) + (pr[i] & 7)] = h;
        }
        __syncthreads();

        if (kt + 2 < KT)   // -> global_prefetch_b8 two tiles ahead
            __builtin_prefetch(&X[(size_t)(kt * 32 + 64) * N + n0], 0, 1);

        // issue next B tile's global loads before the WMMA burst
        if (kt + 1 < KT) {
            #pragma unroll
            for (int i = 0; i < 8; ++i) {
                size_t krow = (size_t)(kt + 1) * 32 + 2 * pr[i];
                blo[i] = X[krow * N + n0 + colr[i]];
                bhi[i] = X[(krow + 1) * N + n0 + colr[i]];
            }
        }

        // one clause of A-fragment loads, then uninterrupted WMMA burst
        v16h bf = lbF[kt & 1][lane];
        v16h af[MW];
        #pragma unroll
        for (int w = 0; w < MW; ++w)
            af[w] = *(const v16h*)(wp + ((size_t)w * KT + kt) * 512);
        #pragma unroll
        for (int w = 0; w < MW; ++w)
            acc[w] = __builtin_amdgcn_wmma_f32_16x16x32_f16(
                false, af[w], false, bf, (short)0, acc[w], false, false);
        // no trailing barrier: next store targets the other LDS buffer
    }

    #pragma unroll
    for (int w = 0; w < MW; ++w) {
        int mt = mt0 + w;
        if (mt >= MT) break;
        #pragma unroll
        for (int q = 0; q < 8; ++q) {
            int m = mt * 16 + q + 8 * hf;
            int n = n0 + rc;
            if (m < M) {
                float r = acc[w][q];
                if (bias) r += bias[m];
                if (flags & 1) r = fmaxf(r, 0.f);
                if (flags & 2) out[(size_t)n * M + m] = r;
                else           out[(size_t)m * N + n] = r;
            }
        }
    }
}

// =====================================================================
// Projection scatter: pf[bv, :, i3d[i]] = conv[bv, :, i2d[i]]
// =====================================================================
__global__ void scatter_proj_kernel(const float* __restrict__ conv,
                                    const int* __restrict__ i3d,
                                    const int* __restrict__ i2d,
                                    float* __restrict__ pf)
{
    int t = blockIdx.x * blockDim.x + threadIdx.x;
    if (t >= kBV * kN) return;
    int bv = t / kN, i = t % kN;
    int a3 = i3d[bv * (kN + 1) + 1 + i];
    int a2 = i2d[bv * (kN + 1) + 1 + i];
    const float* src = conv + (size_t)bv * kCimg * kHW;
    float*       dst = pf   + (size_t)bv * kCimg * kN;
    #pragma unroll
    for (int c = 0; c < kCimg; ++c)
        dst[(size_t)c * kN + a3] = src[(size_t)c * kHW + a2];
}

// max over the V views: imgcn[b,c,n] = max_v pf[b*V+v, c, n]
__global__ void maxv_kernel(const float* __restrict__ pf, float* __restrict__ imgcn)
{
    int t = blockIdx.x * blockDim.x + threadIdx.x;
    if (t >= kB * kCimg * kN) return;
    int b = t / (kCimg * kN), r = t % (kCimg * kN);
    float m = pf[((size_t)b * kV) * kCimg * kN + r];
    for (int v = 1; v < kV; ++v)
        m = fmaxf(m, pf[((size_t)(b * kV + v)) * kCimg * kN + r]);
    imgcn[t] = m;
}

// =====================================================================
// Farthest point sampling: one block per batch, LDS argmax reduction
// =====================================================================
__global__ void fps_kernel(const float* __restrict__ xyz, int* __restrict__ fi,
                           int N, int S)
{
    constexpr int T = 256;
    int b   = blockIdx.x;
    int tid = threadIdx.x;
    const float* p = xyz + (size_t)b * N * 3;
    int* out = fi + (size_t)b * S;

    __shared__ float sd[T];
    __shared__ int   si[T];
    __shared__ int   s_last;

    float dist[32];            // N/T = 8192/256
    int per = N / T;
    for (int i = 0; i < per; ++i) dist[i] = 1e10f;
    if (tid == 0) { out[0] = 0; s_last = 0; }
    __syncthreads();

    for (int it = 1; it < S; ++it) {
        int last = s_last;
        float lx = p[last * 3], ly = p[last * 3 + 1], lz = p[last * 3 + 2];
        float best = -1.f; int bi = 0;
        for (int i = 0; i < per; ++i) {
            int n = tid + i * T;
            float dx = p[n * 3] - lx, dy = p[n * 3 + 1] - ly, dz = p[n * 3 + 2] - lz;
            float d = dx * dx + dy * dy + dz * dz;
            if (d < dist[i]) dist[i] = d;
            if (dist[i] > best) { best = dist[i]; bi = n; }
        }
        sd[tid] = best; si[tid] = bi;
        __syncthreads();
        for (int st = T / 2; st > 0; st >>= 1) {
            if (tid < st && sd[tid + st] > sd[tid]) {
                sd[tid] = sd[tid + st]; si[tid] = si[tid + st];
            }
            __syncthreads();
        }
        if (tid == 0) { out[it] = si[0]; s_last = si[0]; }
        __syncthreads();
    }
}

// gather sampled centers: nxyz[b,s,:] = xyz[b, fi[b,s], :]
__global__ void gather_pts_kernel(const float* __restrict__ xyz,
                                  const int* __restrict__ fi,
                                  float* __restrict__ nxyz, int N, int S, int total)
{
    int t = blockIdx.x * blockDim.x + threadIdx.x;
    if (t >= total) return;
    int b = t / S;
    int n = fi[t];
    #pragma unroll
    for (int c = 0; c < 3; ++c)
        nxyz[(size_t)t * 3 + c] = xyz[(size_t)b * N * 3 + (size_t)n * 3 + c];
}

// ball grouping: first ns in-radius indices (in index order), pad with first
__global__ void ball_group_kernel(const float* __restrict__ xyz,
                                  const float* __restrict__ nxyz,
                                  int* __restrict__ gi,
                                  int N, int S, int ns, float r2, int total)
{
    int t = blockIdx.x * blockDim.x + threadIdx.x;
    if (t >= total) return;
    int b = t / S;
    const float* p = xyz + (size_t)b * N * 3;
    const float* c = nxyz + (size_t)t * 3;
    int* g = gi + (size_t)t * ns;
    float cx = c[0], cy = c[1], cz = c[2];
    int cnt = 0;
    for (int n = 0; n < N && cnt < ns; ++n) {
        float dx = p[n * 3] - cx, dy = p[n * 3 + 1] - cy, dz = p[n * 3 + 2] - cz;
        if (dx * dx + dy * dy + dz * dz < r2) g[cnt++] = n;
    }
    int f = (cnt > 0) ? g[0] : 0;
    for (; cnt < ns; ++cnt) g[cnt] = f;
}

// grouped relative coords -> X (3, S*ns)
__global__ void gather_geom_kernel(const float* __restrict__ xyz,
                                   const float* __restrict__ nxyz,
                                   const int* __restrict__ gi,
                                   float* __restrict__ X, int S, int ns)
{
    int t = blockIdx.x * blockDim.x + threadIdx.x;
    int total = S * ns;
    if (t >= total) return;
    int s = t / ns;
    int n = gi[t];
    #pragma unroll
    for (int c = 0; c < 3; ++c)
        X[(size_t)c * total + t] = xyz[(size_t)n * 3 + c] - nxyz[(size_t)s * 3 + c];
}

// grouped features -> X (C, S*ns) from channel-major (C,N) features
__global__ void gather_feat_kernel(const float* __restrict__ fcn,
                                   const int* __restrict__ gi,
                                   float* __restrict__ X, int S, int ns, int C, int N)
{
    int t = blockIdx.x * blockDim.x + threadIdx.x;
    int sn = S * ns;
    if (t >= C * sn) return;
    int c = t / sn, e = t % sn;
    X[t] = fcn[(size_t)c * N + gi[e]];
}

// max over ns:  out[m, s] = max_j X[m, s*ns + j]   (out pre-offset, ld = S)
__global__ void maxpool_ns_kernel(const float* __restrict__ X, float* __restrict__ out,
                                  int M, int S, int ns)
{
    int t = blockIdx.x * blockDim.x + threadIdx.x;
    if (t >= M * S) return;
    int m = t / S, s = t % S;
    const float* p = X + (size_t)m * S * ns + (size_t)s * ns;
    float r = p[0];
    for (int j = 1; j < ns; ++j) r = fmaxf(r, p[j]);
    out[(size_t)m * S + s] = r;
}

// 3-NN in the fused 6-d coords (both halves identical -> d2_6 = 2*d2_3)
__global__ void knn3_kernel(const float* __restrict__ unk, const float* __restrict__ kn,
                            int n_unk, int m, int* __restrict__ idx3,
                            float* __restrict__ w3)
{
    int n = blockIdx.x * blockDim.x + threadIdx.x;
    if (n >= n_unk) return;
    float ux = unk[n * 3], uy = unk[n * 3 + 1], uz = unk[n * 3 + 2];
    float d0 = 1e30f, d1 = 1e30f, d2 = 1e30f;
    int   i0 = 0, i1 = 0, i2 = 0;
    for (int j = 0; j < m; ++j) {
        float dx = kn[j * 3] - ux, dy = kn[j * 3 + 1] - uy, dz = kn[j * 3 + 2] - uz;
        float d = 2.f * (dx * dx + dy * dy + dz * dz);
        if (d < d0)      { d2 = d1; i2 = i1; d1 = d0; i1 = i0; d0 = d; i0 = j; }
        else if (d < d1) { d2 = d1; i2 = i1; d1 = d; i1 = j; }
        else if (d < d2) { d2 = d; i2 = j; }
    }
    float w0 = 1.f / (d0 + 1e-8f), w1 = 1.f / (d1 + 1e-8f), w2 = 1.f / (d2 + 1e-8f);
    float s = w0 + w1 + w2;
    idx3[n * 3] = i0; idx3[n * 3 + 1] = i1; idx3[n * 3 + 2] = i2;
    w3[n * 3] = w0 / s; w3[n * 3 + 1] = w1 / s; w3[n * 3 + 2] = w2 / s;
}

// FP0 input: X[0:192,n] = interp(known), X[192:224,n] = image features
__global__ void interp_concat_kernel(const float* __restrict__ kn,
                                     const float* __restrict__ imgcn_b,
                                     const int* __restrict__ idx3,
                                     const float* __restrict__ w3,
                                     float* __restrict__ X, int N, int S,
                                     int CK, int CI)
{
    int t = blockIdx.x * blockDim.x + threadIdx.x;
    if (t >= (CK + CI) * N) return;
    int c = t / N, n = t % N;
    float r;
    if (c < CK) {
        const int*   id = idx3 + n * 3;
        const float* w  = w3 + n * 3;
        const float* kc = kn + (size_t)c * S;
        r = kc[id[0]] * w[0] + kc[id[1]] * w[1] + kc[id[2]] * w[2];
    } else {
        r = imgcn_b[(size_t)(c - CK) * N + n];
    }
    X[(size_t)c * N + n] = r;
}

// =====================================================================
// Host orchestration
// =====================================================================
struct Layer  { const float *w, *g, *b; int cin, cout; };
struct Packed { const _Float16* p; const float* bias; int M, MT, KT; };

extern "C" void kernel_launch(void* const* d_in, const int* in_sizes, int n_in,
                              void* d_out, int out_size, void* d_ws, size_t ws_size,
                              hipStream_t stream)
{
    (void)in_sizes; (void)n_in; (void)out_size; (void)ws_size;

    const float* pc  = (const float*)d_in[0];   // (B, N, 3)
    const float* imf = (const float*)d_in[1];   // (BV, 128, H, W)
    const int*   i3d = (const int*)d_in[2];     // (BV, N+1)
    const int*   i2d = (const int*)d_in[3];     // (BV, N+1)

    // ---- walk params in reference insertion order ----
    int cur = 4;
    auto nxtp = [&]() { return (const float*)d_in[cur++]; };
    const float* conv_w = nxtp();               // (32,128)
    const float* conv_b = nxtp();               // (32,)

    const int mlps[2][3] = {{16, 16, 32}, {32, 32, 64}};
    Layer geomL[2][3], featL[2][3];
    for (int lvl = 0; lvl < 4; ++lvl)           // sa_geom (only level 0 live)
        for (int s = 0; s < 2; ++s) {
            int cin = 3;
            for (int l = 0; l < 3; ++l) {
                const float* w = nxtp(); const float* g = nxtp(); const float* b = nxtp();
                if (lvl == 0) { geomL[s][l] = {w, g, b, cin, mlps[s][l]}; cin = mlps[s][l]; }
            }
        }
    for (int lvl = 0; lvl < 4; ++lvl)           // sa_feat (only level 0 live)
        for (int s = 0; s < 2; ++s) {
            int cin = 32;
            for (int l = 0; l < 3; ++l) {
                const float* w = nxtp(); const float* g = nxtp(); const float* b = nxtp();
                if (lvl == 0) { featL[s][l] = {w, g, b, cin, mlps[s][l]}; cin = mlps[s][l]; }
            }
        }
    Layer fp0L[2];
    {
        const int fpch[3] = {224, 128, 128};
        for (int i = 0; i < 4; ++i)             // fp (only fp[0] live)
            for (int l = 0; l < 2; ++l) {
                const float* w = nxtp(); const float* g = nxtp(); const float* b = nxtp();
                if (i == 0) fp0L[l] = {w, g, b, fpch[l], fpch[l + 1]};
            }
    }
    Layer clsL[2];
    {
        const int ch[3] = {128, 128, kNCLS};
        for (int l = 0; l < 2; ++l) {
            const float* w = nxtp(); const float* g = nxtp(); const float* b = nxtp();
            clsL[l] = {w, g, b, ch[l], ch[l + 1]};
        }
    }

    // ---- workspace carving ----
    char* wsb = (char*)d_ws;
    size_t off = 0;
    auto alloc = [&](size_t bytes) {
        void* p = wsb + off;
        off += (bytes + 255) & ~(size_t)255;
        return p;
    };
    float* convOut = (float*)alloc((size_t)kBV * kCimg * kHW * 4);
    float* pf      = (float*)alloc((size_t)kBV * kCimg * kN * 4);
    float* imgcn   = (float*)alloc((size_t)kB * kCimg * kN * 4);
    int*   fi      = (int*)  alloc((size_t)kB * kS * 4);
    float* nxyz    = (float*)alloc((size_t)kB * kS * 3 * 4);
    int*   gi0     = (int*)  alloc((size_t)kB * kS * 16 * 4);
    int*   gi1     = (int*)  alloc((size_t)kB * kS * 32 * 4);
    float* knf     = (float*)alloc((size_t)kB * 192 * kS * 4);
    int*   idx3    = (int*)  alloc((size_t)kB * kN * 3 * 4);
    float* w3      = (float*)alloc((size_t)kB * kN * 3 * 4);
    const size_t PING = (size_t)64 * 32768;   // >= 224*8192 as well
    float* pingA   = (float*)alloc(PING * 4);
    float* pingB   = (float*)alloc(PING * 4);

    // ---- weight packing (BN scale folded into f16 fragments) ----
    auto pack = [&](const Layer& L, bool bn) -> Packed {
        int MT = ceilDiv(L.cout, 16), KT = ceilDiv(L.cin, 32);
        int total = MT * KT * 512;
        _Float16* dst = (_Float16*)alloc((size_t)total * 2);
        pack_w_kernel<<<ceilDiv(total, 256), 256, 0, stream>>>(
            L.w, bn ? L.g : nullptr, kBNS, dst, L.cout, L.cin, KT, total);
        return {dst, L.b, L.cout, MT, KT};
    };
    Packed pkConv = pack(Layer{conv_w, nullptr, conv_b, 128, 32}, false);
    Packed pkGeom[2][3], pkFeat[2][3], pkFp[2], pkCls[2];
    for (int s = 0; s < 2; ++s)
        for (int l = 0; l < 3; ++l) { pkGeom[s][l] = pack(geomL[s][l], true);
                                      pkFeat[s][l] = pack(featL[s][l], true); }
    for (int l = 0; l < 2; ++l) { pkFp[l] = pack(fp0L[l], true);
                                  pkCls[l] = pack(clsL[l], true); }

    // ---- GEMM launcher: picks MW, zero-pads ragged K rows of X ----
    auto gemm = [&](const Packed& P, float* X, float* out, int N, int K,
                    int flags, int batch, long xs, long os) {
        if (K & 31)   // zero the K..KT*32 pad rows (deterministic each call)
            hipMemsetAsync(X + (size_t)K * N, 0,
                           (size_t)(P.KT * 32 - K) * N * 4, stream);
        int MW = (P.MT % 4 == 0) ? 4 : ((P.MT % 2 == 0) ? 2 : 1);
        dim3 grid(N / 16, ceilDiv(P.MT, MW), batch);
        if (MW == 4)
            gemm_packed<4><<<grid, 32, 0, stream>>>(P.p, X, P.bias, out,
                P.M, N, P.KT, P.MT, flags, xs, os);
        else if (MW == 2)
            gemm_packed<2><<<grid, 32, 0, stream>>>(P.p, X, P.bias, out,
                P.M, N, P.KT, P.MT, flags, xs, os);
        else
            gemm_packed<1><<<grid, 32, 0, stream>>>(P.p, X, P.bias, out,
                P.M, N, P.KT, P.MT, flags, xs, os);
    };

    // 1) image 1x1 conv, batched over BV (K=128 aligned, input untouched)
    gemm(pkConv, (float*)imf, convOut, kHW, 128, 0,
         kBV, (long)128 * kHW, (long)32 * kHW);

    // 2) projection scatter + max over views
    hipMemsetAsync(pf, 0, (size_t)kBV * kCimg * kN * 4, stream);
    scatter_proj_kernel<<<ceilDiv(kBV * kN, 256), 256, 0, stream>>>(convOut, i3d, i2d, pf);
    maxv_kernel<<<ceilDiv(kB * kCimg * kN, 256), 256, 0, stream>>>(pf, imgcn);

    // 3) FPS + centers + ball groups (shared by both branches)
    fps_kernel<<<kB, 256, 0, stream>>>(pc, fi, kN, kS);
    gather_pts_kernel<<<ceilDiv(kB * kS, 256), 256, 0, stream>>>(pc, fi, nxyz, kN, kS, kB * kS);
    ball_group_kernel<<<ceilDiv(kB * kS, 256), 256, 0, stream>>>(pc, nxyz, gi0, kN, kS, 16,
                                                                 0.05f * 0.05f, kB * kS);
    ball_group_kernel<<<ceilDiv(kB * kS, 256), 256, 0, stream>>>(pc, nxyz, gi1, kN, kS, 32,
                                                                 0.1f * 0.1f, kB * kS);

    const int nsArr[2] = {16, 32};
    int* giArr[2] = {gi0, gi1};
    const int choffGeom[2] = {0, 32};
    const int choffFeat[2] = {96, 128};

    for (int b = 0; b < kB; ++b) {
        const float* pc_b    = pc    + (size_t)b * kN * 3;
        const float* imgcn_b = imgcn + (size_t)b * kCimg * kN;
        const float* nxyz_b  = nxyz  + (size_t)b * kS * 3;
        float*       kn_b    = knf   + (size_t)b * 192 * kS;

        // 4) SA level 0, both scales, both branches
        for (int s = 0; s < 2; ++s) {
            int ns = nsArr[s];
            int SN = kS * ns;
            const int* gi_b = giArr[s] + (size_t)b * kS * ns;

            // geometry branch (use_xyz): relative coords (3, S*ns)
            gather_geom_kernel<<<ceilDiv(SN, 256), 256, 0, stream>>>(pc_b, nxyz_b, gi_b,
                                                                     pingA, kS, ns);
            {
                float *in = pingA, *outp = pingB;
                for (int l = 0; l < 3; ++l) {
                    gemm(pkGeom[s][l], in, outp, SN, geomL[s][l].cin, 1, 1, 0, 0);
                    float* t = in; in = outp; outp = t;
                }
                maxpool_ns_kernel<<<ceilDiv(geomL[s][2].cout * kS, 256), 256, 0, stream>>>(
                    in, kn_b + (size_t)choffGeom[s] * kS, geomL[s][2].cout, kS, ns);
            }

            // feature branch (use_xyz=False): grouped image feats (32, S*ns)
            gather_feat_kernel<<<ceilDiv(kCimg * SN, 256), 256, 0, stream>>>(
                imgcn_b, gi_b, pingA, kS, ns, kCimg, kN);
            {
                float *in = pingA, *outp = pingB;
                for (int l = 0; l < 3; ++l) {
                    gemm(pkFeat[s][l], in, outp, SN, featL[s][l].cin, 1, 1, 0, 0);
                    float* t = in; in = outp; outp = t;
                }
                maxpool_ns_kernel<<<ceilDiv(featL[s][2].cout * kS, 256), 256, 0, stream>>>(
                    in, kn_b + (size_t)choffFeat[s] * kS, featL[s][2].cout, kS, ns);
            }
        }

        // 5) FP0: 3-NN interpolate to 8192 pts, concat image channels
        knn3_kernel<<<ceilDiv(kN, 256), 256, 0, stream>>>(pc_b, nxyz_b, kN, kS,
                                                          idx3 + (size_t)b * kN * 3,
                                                          w3 + (size_t)b * kN * 3);
        interp_concat_kernel<<<ceilDiv(224 * kN, 256), 256, 0, stream>>>(
            kn_b, imgcn_b, idx3 + (size_t)b * kN * 3, w3 + (size_t)b * kN * 3,
            pingA, kN, kS, 192, 32);

        gemm(pkFp[0], pingA, pingB, kN, 224, 1, 1, 0, 0);
        gemm(pkFp[1], pingB, pingA, kN, 128, 1, 1, 0, 0);

        // 6) classifier head; final layer stores transposed -> (N, 21)
        gemm(pkCls[0], pingA, pingB, kN, 128, 1, 1, 0, 0);
        gemm(pkCls[1], pingB, (float*)d_out + (size_t)b * kN * kNCLS,
             kN, 128, 2, 1, 0, 0);
    }
}
